// SpaceTimeMultiHeadAttention_1022202217190
// MI455X (gfx1250) — matrix operand order repaired and checked by reading, compiled, not run on hardware
//
#include <hip/hip_runtime.h>

typedef _Float16 v16h __attribute__((ext_vector_type(16)));
typedef _Float16 v8h  __attribute__((ext_vector_type(8)));
typedef float    v8f  __attribute__((ext_vector_type(8)));

#define DEV __device__ __forceinline__

// ---------------------------------------------------------------------------
// WMMA helper: D = A(16x32 f16) * B(32x16 f16) + C(16x16 f32)
// ---------------------------------------------------------------------------
DEV v8f wmma32(v16h a, v16h b, v8f c) {
  return __builtin_amdgcn_wmma_f32_16x16x32_f16(
      /*neg_a=*/false, a, /*neg_b=*/false, b,
      /*c_mod=*/(short)0, c, /*reuse_a=*/false, /*reuse_b=*/false);
}

DEV v16h cat8(v8h lo, v8h hi) {
  return __builtin_shufflevector(lo, hi, 0, 1, 2, 3, 4, 5, 6, 7,
                                 8, 9, 10, 11, 12, 13, 14, 15);
}

// A-fragment (16x32, MxK) from a row-major f16 tile (stride in halves).
// Lanes 0-15 = M rows; chunk0 K=16g..16g+7, chunk1 K=16g+8..16g+15.
DEV v16h load_a(const _Float16* base, int stride, int ln, int g) {
  v8h lo = *(const v8h*)(base + (size_t)ln * stride + 16 * g);
  v8h hi = *(const v8h*)(base + (size_t)ln * stride + 16 * g + 8);
  return cat8(lo, hi);
}

// B-fragment (32x16, KxN) from a K-contiguous (N-major) f16 tile:
// lane holds N = ln, contraction K = 16g..16g+15 contiguous.
DEV v16h load_b(const _Float16* base, int stride, int ln, int g) {
  return *(const v16h*)(base + (size_t)ln * stride + 16 * g);
}

// Max-reduce across each 16-lane half using VALU DPP ROW_XMASK (no LDS).
DEV float rowmax16(float x) {
  int t;
  t = __builtin_amdgcn_update_dpp(0, __float_as_int(x), 0x161, 0xF, 0xF, true);
  x = fmaxf(x, __int_as_float(t));
  t = __builtin_amdgcn_update_dpp(0, __float_as_int(x), 0x162, 0xF, 0xF, true);
  x = fmaxf(x, __int_as_float(t));
  t = __builtin_amdgcn_update_dpp(0, __float_as_int(x), 0x164, 0xF, 0xF, true);
  x = fmaxf(x, __int_as_float(t));
  t = __builtin_amdgcn_update_dpp(0, __float_as_int(x), 0x168, 0xF, 0xF, true);
  x = fmaxf(x, __int_as_float(t));
  return x;
}

// ---------------------------------------------------------------------------
// Shared GEMM body: Out[M,512-seg] = A[M,512] * W[512,512] + bias, times alpha.
// Block tile 128x128, BK=32, 8 waves, wave tile 32x64.
// ---------------------------------------------------------------------------
template <bool AH, bool OH>
DEV void gemm_body(_Float16* As, _Float16* Bs, const void* Aptr,
                   const float* W, const float* bias, void* Out, float alpha,
                   int m0, int nl) {
  const int tid = threadIdx.x;
  const int lane = tid & 31, wid = tid >> 5;
  const int ln = lane & 15, g = lane >> 4;
  const int wm = wid & 3, wn = wid >> 2;            // 4 x 2 wave grid

  v8f acc[2][4] = {};
  const int arow = tid >> 1, ac0 = (tid & 1) * 16;  // A: 16 halves/thread
  const int bk = tid >> 3, bn0 = (tid & 7) * 16;    // B: one k-row, 16 n

  for (int k0 = 0; k0 < 512; k0 += 32) {
    __syncthreads();
    // ---- stage A tile [128m][32k], f32->f16 convert if needed ----
    {
      _Float16* dst = As + arow * 32 + ac0;
      if constexpr (AH) {
        const _Float16* src =
            (const _Float16*)Aptr + (size_t)(m0 + arow) * 512 + k0 + ac0;
        *(v8h*)dst = *(const v8h*)src;
        *(v8h*)(dst + 8) = *(const v8h*)(src + 8);
      } else {
        const float* src =
            (const float*)Aptr + (size_t)(m0 + arow) * 512 + k0 + ac0;
#pragma unroll
        for (int j = 0; j < 16; j += 4) {
          float4 f = *(const float4*)(src + j);
          dst[j + 0] = (_Float16)f.x;
          dst[j + 1] = (_Float16)f.y;
          dst[j + 2] = (_Float16)f.z;
          dst[j + 3] = (_Float16)f.w;
        }
      }
    }
    // ---- stage W tile transposed Bs[n][k]; coalesced global reads ----
    {
      const float* src = W + (size_t)(k0 + bk) * 512 + nl + bn0;
#pragma unroll
      for (int j = 0; j < 16; j += 4) {
        float4 f = *(const float4*)(src + j);
        Bs[(bn0 + j + 0) * 32 + bk] = (_Float16)f.x;
        Bs[(bn0 + j + 1) * 32 + bk] = (_Float16)f.y;
        Bs[(bn0 + j + 2) * 32 + bk] = (_Float16)f.z;
        Bs[(bn0 + j + 3) * 32 + bk] = (_Float16)f.w;
      }
    }
    __syncthreads();

    v16h af[2], bf[4];
#pragma unroll
    for (int mt = 0; mt < 2; ++mt)
      af[mt] = load_a(As + (wm * 32 + mt * 16) * 32, 32, ln, g);
#pragma unroll
    for (int nt = 0; nt < 4; ++nt)
      bf[nt] = load_b(Bs + (wn * 64 + nt * 16) * 32, 32, ln, g);
#pragma unroll
    for (int mt = 0; mt < 2; ++mt)
#pragma unroll
      for (int nt = 0; nt < 4; ++nt)
        acc[mt][nt] = wmma32(af[mt], bf[nt], acc[mt][nt]);
  }

  // ---- epilogue: C layout row = r + 8g, col = ln ----
#pragma unroll
  for (int mt = 0; mt < 2; ++mt) {
#pragma unroll
    for (int nt = 0; nt < 4; ++nt) {
      const int col = nl + wn * 64 + nt * 16 + ln;
      const float bb = bias[col];
#pragma unroll
      for (int r = 0; r < 8; ++r) {
        const int row = m0 + wm * 32 + mt * 16 + r + 8 * g;
        const float val = (acc[mt][nt][r] + bb) * alpha;
        if constexpr (OH)
          ((_Float16*)Out)[(size_t)row * 512 + col] = (_Float16)val;
        else
          ((float*)Out)[(size_t)row * 512 + col] = val;
      }
    }
  }
}

struct QKVParams {
  const float* w[3];
  const float* b[3];
  _Float16* out[3];
  float alpha[3];
};

// Fused Q/K/V projection: X is streamed through HBM exactly once.
// Grid: (1536/128 = 12, 32768/128 = 256). Each n-block picks its segment.
__global__ __launch_bounds__(256) void qkv_gemm_k(const float* __restrict__ X,
                                                  QKVParams p) {
  __shared__ __align__(32) _Float16 As[128 * 32];
  __shared__ __align__(32) _Float16 Bs[128 * 32];
  const int n0 = blockIdx.x * 128;
  const int seg = n0 >> 9, nl = n0 & 511;
  gemm_body<false, true>(As, Bs, X, p.w[seg], p.b[seg], p.out[seg],
                         p.alpha[seg], blockIdx.y * 128, nl);
}

// Output projection: attn(f16) @ wo + bo -> f32.
__global__ __launch_bounds__(256) void out_gemm_k(
    const _Float16* __restrict__ A, const float* __restrict__ W,
    const float* __restrict__ bias, void* __restrict__ Out) {
  __shared__ __align__(32) _Float16 As[128 * 32];
  __shared__ __align__(32) _Float16 Bs[128 * 32];
  gemm_body<true, false>(As, Bs, A, W, bias, Out, 1.0f, blockIdx.y * 128,
                         blockIdx.x * 128);
}

// ---------------------------------------------------------------------------
// Flash attention over the L axis; q carries the 1/sqrt(64) scale.
// Block = one (b,t,h) + 128 q-rows; 8 waves x 16 rows. 32 keys per step.
// Row max via DPP; row sum via an extra all-ones WMMA column block.
// ---------------------------------------------------------------------------
__global__ __launch_bounds__(256) void flash_attn_k(
    const _Float16* __restrict__ q, const _Float16* __restrict__ k,
    const _Float16* __restrict__ v, _Float16* __restrict__ attn) {
  __shared__ __align__(32) _Float16 Ks[32 * 64];      // [key][d] row-major
  __shared__ __align__(32) _Float16 VT[64 * 32];      // [d][key] transposed V
  __shared__ __align__(32) _Float16 Pb[8 * 16 * 32];  // per-wave P staging

  const int tid = threadIdx.x;
  const int lane = tid & 31, wid = tid >> 5;
  const int ln = lane & 15, g = lane >> 4;

  const int qtile = blockIdx.x & 3;  // 4 q-tiles of 128 rows
  const int bth = blockIdx.x >> 2;   // 0..511
  const int h = bth & 7;
  const int bt = bth >> 3;           // b*T + t

  const size_t rowQ = (size_t)bt * 512 + qtile * 128 + wid * 16;
  const _Float16* qp = q + rowQ * 512 + h * 64;

  // Resident Q fragments: 16 rows x 64 d = two 16x32 A-fragments.
  const v16h aq0 = load_a(qp, 512, ln, g);
  const v16h aq1 = load_a(qp + 32, 512, ln, g);

  v16h ones;
#pragma unroll
  for (int i = 0; i < 16; ++i) ones[i] = (_Float16)1.0f;

  v8f o[5] = {};  // o[0..3] = output tiles, o[4] = running row sums
  float m_i[8];
#pragma unroll
  for (int r = 0; r < 8; ++r) m_i[r] = -1e30f;

  const int skey = (tid * 8) >> 6;  // cooperative K/V staging indices
  const int sd = (tid * 8) & 63;
  _Float16* pw = Pb + wid * (16 * 32);

  for (int kt = 0; kt < 512; kt += 32) {
    __syncthreads();  // all waves done reading previous Ks/VT
    {
      const size_t srow = ((size_t)bt * 512 + kt + skey) * 512 + h * 64 + sd;
      const v8h kk = *(const v8h*)(k + srow);
      const v8h vv = *(const v8h*)(v + srow);
      *(v8h*)(Ks + skey * 64 + sd) = kk;  // straight copy (row-major)
#pragma unroll
      for (int j = 0; j < 8; ++j) VT[(sd + j) * 32 + skey] = vv[j];  // transpose
    }
    __syncthreads();

    // ---- S = Q * K^T : B-fragments from LDS (rows of K = cols of K^T) ----
    const v16h b00 = load_b(Ks, 64, ln, g);             // keys 0-15,  d 0-31
    const v16h b01 = load_b(Ks + 32, 64, ln, g);        // keys 0-15,  d 32-63
    const v16h b10 = load_b(Ks + 16 * 64, 64, ln, g);   // keys 16-31, d 0-31
    const v16h b11 = load_b(Ks + 16 * 64 + 32, 64, ln, g);
    v8f s0 = {}, s1 = {};
    s0 = wmma32(aq0, b00, s0);
    s0 = wmma32(aq1, b01, s0);
    s1 = wmma32(aq0, b10, s1);
    s1 = wmma32(aq1, b11, s1);

    // ---- online softmax: row r+8g lives across one 16-lane half ----
#pragma unroll
    for (int r = 0; r < 8; ++r) {
      const float a = s0[r], bb = s1[r];
      const float mx = rowmax16(fmaxf(a, bb));
      const float mn = fmaxf(m_i[r], mx);
      const float corr = __expf(m_i[r] - mn);
      m_i[r] = mn;
      const float p0 = __expf(a - mn), p1 = __expf(bb - mn);
#pragma unroll
      for (int t = 0; t < 5; ++t) o[t][r] = o[t][r] * corr;
      const int prow = r + 8 * g;  // C-layout -> row-major P in LDS
      pw[prow * 32 + ln] = (_Float16)p0;
      pw[prow * 32 + 16 + ln] = (_Float16)p1;
    }
    __builtin_amdgcn_wave_barrier();  // keep per-wave DS store->load order

    // ---- O += P(16x32) * V(32x64); row sums via all-ones B tile ----
    const v16h ap = load_a(pw, 32, ln, g);
#pragma unroll
    for (int t = 0; t < 4; ++t) {
      const v16h bv = load_b(VT + (16 * t) * 32, 32, ln, g);
      o[t] = wmma32(ap, bv, o[t]);
    }
    o[4] = wmma32(ap, ones, o[4]);  // every column = rowsum(P)
  }

  // ---- finalize: divide by row sums, store f16 ----
  _Float16* op = attn + rowQ * 512 + h * 64;
#pragma unroll
  for (int r = 0; r < 8; ++r) {
    const float inv = 1.0f / o[4][r];
#pragma unroll
    for (int t = 0; t < 4; ++t)
      op[(size_t)(r + 8 * g) * 512 + 16 * t + ln] = (_Float16)(o[t][r] * inv);
  }
}

// ---------------------------------------------------------------------------
extern "C" void kernel_launch(void* const* d_in, const int* in_sizes, int n_in,
                              void* d_out, int out_size, void* d_ws,
                              size_t ws_size, hipStream_t stream) {
  (void)in_sizes; (void)n_in; (void)out_size; (void)ws_size;
  const float* x  = (const float*)d_in[0];
  const float* wq = (const float*)d_in[1];
  const float* bq = (const float*)d_in[2];
  const float* wk = (const float*)d_in[3];
  const float* bk = (const float*)d_in[4];
  const float* wv = (const float*)d_in[5];
  const float* bv = (const float*)d_in[6];
  const float* wo = (const float*)d_in[7];
  const float* bo = (const float*)d_in[8];

  const size_t elems = (size_t)32768 * 512;
  _Float16* qh = (_Float16*)d_ws;
  _Float16* kh = qh + elems;
  _Float16* vh = kh + elems;
  _Float16* ah = vh + elems;  // 134 MB f16 scratch total

  QKVParams p;
  p.w[0] = wq; p.w[1] = wk; p.w[2] = wv;
  p.b[0] = bq; p.b[1] = bk; p.b[2] = bv;
  p.out[0] = qh; p.out[1] = kh; p.out[2] = vh;
  p.alpha[0] = 0.125f; p.alpha[1] = 1.0f; p.alpha[2] = 1.0f;

  // Fused QKV projection: X read once. Grid (1536/128, 32768/128).
  qkv_gemm_k<<<dim3(12, 256), 256, 0, stream>>>(x, p);
  // Attention: B*T*H * 4 q-tiles = 2048 blocks.
  flash_attn_k<<<2048, 256, 0, stream>>>(qh, kh, vh, ah);
  // Output projection -> f32 d_out. Grid (512/128, 32768/128).
  out_gemm_k<<<dim3(4, 256), 256, 0, stream>>>(ah, wo, bo, d_out);
}
